// SelfAttention_71476845740490
// MI455X (gfx1250) — matrix operand order repaired and checked by reading, not compiled, not run on hardware
//
#include <hip/hip_runtime.h>

// Self-attention (B=16, S=2048, Din=512, Dout=64), flash-attention style.
// Pass 1: Q/K/V = X @ W[0..2] in f16 (workspace, 12.6 MB, fits 192MB L2).
// Pass 2: per 16-row query tile: online-softmax attention with
//         v_wmma_f32_16x16x32_f16 for both Q*K^T and P*V.
// K tiles are staged global->LDS with CDNA5 async-DMA
// (global_load_async_to_lds_b128 + s_wait_asynccnt); next tile prefetched
// with global_prefetch_b8.

typedef __attribute__((ext_vector_type(16))) _Float16 v16h;
typedef __attribute__((ext_vector_type(8)))  _Float16 v8h;
typedef __attribute__((ext_vector_type(8)))  float    v8f;
typedef __attribute__((ext_vector_type(4)))  float    v4f;

#define BATCH 16
#define SEQ   2048
#define DIN   512
#define DOUT  64
#define BS    (BATCH*SEQ)

static __device__ __forceinline__ v16h make_a16(v8h lo, v8h hi) {
  v16h a;
  #pragma unroll
  for (int i = 0; i < 8; i++) { a[i] = lo[i]; a[i + 8] = hi[i]; }
  return a;
}

// ---------------------------------------------------------------------------
// Projection: out[m][row][n] = sum_k X[row][k] * W[m][k][n], stored as f16.
// grid = (BS/128, 3), block = 256 (8 waves; each wave -> 16x64 output tile).
// ---------------------------------------------------------------------------
__global__ __launch_bounds__(256) void proj_kernel(
    const float* __restrict__ X,     // [BS][512] f32
    const float* __restrict__ W,     // [1000][512][64] f32 (slices 0..2 used)
    _Float16* __restrict__ out)      // [3][BS][64] f16
{
  __shared__ __align__(16) _Float16 WT[DOUT][DIN + 8];  // transposed weights

  const int tid  = threadIdx.x;
  const int wave = tid >> 5;
  const int lane = tid & 31;
  const int l16  = lane & 15;
  const int hi   = lane >> 4;
  const int m    = blockIdx.y;                     // which weight matrix

  const float*  Wm = W   + (size_t)m * DIN * DOUT;
  _Float16*     Om = out + (size_t)m * BS  * DOUT;

  // Stage W transposed into LDS as f16 (16B coalesced global reads).
  for (int i = tid; i < (DIN * DOUT) / 4; i += 256) {
    int idx = i * 4;
    int k = idx >> 6;
    int n = idx & 63;
    v4f w = *(const v4f*)(Wm + idx);
    #pragma unroll
    for (int e = 0; e < 4; e++) WT[n + e][k] = (_Float16)w[e];
  }
  __syncthreads();

  const int    rowbase = blockIdx.x * 128 + wave * 16;
  const float* Xrow    = X + (size_t)(rowbase + l16) * DIN;

  v8f acc[4];
  #pragma unroll
  for (int n = 0; n < 4; n++) acc[n] = {};

  #pragma unroll 1
  for (int c = 0; c < DIN / 32; c++) {
    // A operand: 16x32 f16 chunk of X. Lane=row; halves h -> k per ISA layout:
    // k = c*32 + (h<8 ? h : h+8) + 8*hi  => two runs of 8 consecutive floats.
    const int k0 = c * 32 + hi * 8;
    v4f f0 = *(const v4f*)(Xrow + k0);
    v4f f1 = *(const v4f*)(Xrow + k0 + 4);
    v4f f2 = *(const v4f*)(Xrow + k0 + 16);
    v4f f3 = *(const v4f*)(Xrow + k0 + 20);
    v16h a;
    #pragma unroll
    for (int i = 0; i < 4; i++) {
      a[i]      = (_Float16)f0[i];
      a[i + 4]  = (_Float16)f1[i];
      a[i + 8]  = (_Float16)f2[i];
      a[i + 12] = (_Float16)f3[i];
    }
    #pragma unroll
    for (int n = 0; n < 4; n++) {
      // B operand: lane holds col n, halves h -> k = c*32 + (h<8 ? h : h+8) + 8*hi
      const _Float16* bp = &WT[n * 16 + l16][c * 32 + hi * 8];
      v16h b = make_a16(*(const v8h*)bp, *(const v8h*)(bp + 16));
      acc[n] = __builtin_amdgcn_wmma_f32_16x16x32_f16(
          false, a, false, b, (short)0, acc[n], false, false);
    }
  }

  // C layout: VGPR r -> row r + 8*hi, col = lane%16 (+16*n).
  #pragma unroll
  for (int n = 0; n < 4; n++) {
    #pragma unroll
    for (int r = 0; r < 8; r++) {
      int row = rowbase + r + 8 * hi;
      Om[(size_t)row * DOUT + n * 16 + l16] = (_Float16)acc[n][r];
    }
  }
}

// ---------------------------------------------------------------------------
// Flash attention. grid = (SEQ/128, BATCH), block = 256 (8 waves).
// Each wave owns a 16-row query tile; block iterates over 128-key tiles.
// ---------------------------------------------------------------------------
__global__ __launch_bounds__(256) void attn_kernel(
    const _Float16* __restrict__ Qh,   // [BS][64]
    const _Float16* __restrict__ Kh,   // [BS][64]
    const _Float16* __restrict__ Vh,   // [BS][64]
    float* __restrict__ Out)           // [BS][64] f32
{
  __shared__ __align__(16) _Float16 Ksh[128][DOUT];      // K tile, row-major
  __shared__ __align__(16) _Float16 VT[DOUT][128 + 8];   // V tile, transposed
  __shared__ __align__(16) _Float16 Pl[8][16][128];      // per-wave P scratch

  const int tid  = threadIdx.x;
  const int wave = tid >> 5;
  const int lane = tid & 31;
  const int l16  = lane & 15;
  const int hi   = lane >> 4;
  const size_t batch_base = (size_t)blockIdx.y * SEQ;
  const int qrow0 = blockIdx.x * 128 + wave * 16;        // within sequence

  // Q tile as two 16x32 A operands (lane=row, halves -> d per A layout).
  const _Float16* Qrow = Qh + (batch_base + qrow0 + l16) * DOUT;
  v16h aq[2];
  #pragma unroll
  for (int c = 0; c < 2; c++) {
    const _Float16* p = Qrow + c * 32 + hi * 8;
    aq[c] = make_a16(*(const v8h*)p, *(const v8h*)(p + 16));
  }

  v8f acc[4];
  #pragma unroll
  for (int n = 0; n < 4; n++) acc[n] = {};
  float mrow[8], lrow[8];
  #pragma unroll
  for (int r = 0; r < 8; r++) { mrow[r] = -__builtin_inff(); lrow[r] = 0.f; }

  #pragma unroll 1
  for (int kt = 0; kt < SEQ / 128; kt++) {
    __syncthreads();   // previous tile fully consumed by all waves
    const _Float16* Kt = Kh + (batch_base + kt * 128) * DOUT;
    const _Float16* Vt = Vh + (batch_base + kt * 128) * DOUT;

    // --- Stage K via CDNA5 async DMA (no VGPR round-trip, ASYNCcnt), and
    // --- V transposed via 16B loads + scattered b16 LDS stores.
    #pragma unroll
    for (int it = 0; it < 4; it++) {
      int j   = tid + 256 * it;        // 16-byte chunk id, 0..1023
      int key = j >> 3;
      int d0  = (j & 7) * 8;
      unsigned lds_addr = (unsigned)(size_t)&Ksh[key][d0];
      const _Float16* gk = Kt + key * DOUT + d0;
      asm volatile("global_load_async_to_lds_b128 %0, %1, off"
                   :: "v"(lds_addr), "v"(gk) : "memory");
      v8h vv = *(const v8h*)(Vt + key * DOUT + d0);
      #pragma unroll
      for (int e = 0; e < 8; e++) VT[d0 + e][key] = vv[e];
    }
    // Prefetch next tile (L2 -> WGP$): one 64B line per thread covers 16KB.
    if (kt + 1 < SEQ / 128) {
      __builtin_prefetch((const char*)(Kt + 128 * DOUT) + tid * 64, 0, 3);
      __builtin_prefetch((const char*)(Vt + 128 * DOUT) + tid * 64, 0, 3);
    }
    asm volatile("s_wait_asynccnt 0x0" ::: "memory");
    __syncthreads();

    // S = (Q * K^T) * 1/sqrt(64). 8 key-subtiles of 16, K-dim 64 = 2 chunks.
    v8f s[8];
    #pragma unroll
    for (int t = 0; t < 8; t++) {
      v8f cc = {};
      #pragma unroll
      for (int c = 0; c < 2; c++) {
        const _Float16* bp = &Ksh[t * 16 + l16][c * 32 + hi * 8];
        v16h bk = make_a16(*(const v8h*)bp, *(const v8h*)(bp + 16));
        cc = __builtin_amdgcn_wmma_f32_16x16x32_f16(
            false, aq[c], false, bk, (short)0, cc, false, false);
      }
      s[t] = cc * 0.125f;
    }

    // Online softmax. Row r+8*hi lives in VGPR r across one 16-lane group.
    float corr[8];
    #pragma unroll
    for (int r = 0; r < 8; r++) {
      float mx = s[0][r];
      #pragma unroll
      for (int t = 1; t < 8; t++) mx = fmaxf(mx, s[t][r]);
      #pragma unroll
      for (int off = 1; off < 16; off <<= 1)
        mx = fmaxf(mx, __shfl_xor(mx, off, 32));
      float mnew = fmaxf(mrow[r], mx);
      corr[r] = __expf(mrow[r] - mnew);
      mrow[r] = mnew;
    }
    #pragma unroll
    for (int t = 0; t < 8; t++) {
      #pragma unroll
      for (int r = 0; r < 8; r++) s[t][r] = __expf(s[t][r] - mrow[r]);
    }
    #pragma unroll
    for (int r = 0; r < 8; r++) {
      float rs = 0.f;
      #pragma unroll
      for (int t = 0; t < 8; t++) rs += s[t][r];
      #pragma unroll
      for (int off = 1; off < 16; off <<= 1) rs += __shfl_xor(rs, off, 32);
      lrow[r] = lrow[r] * corr[r] + rs;
      #pragma unroll
      for (int n = 0; n < 4; n++) acc[n][r] *= corr[r];
    }

    // Transpose P (C layout -> A layout source) through per-wave LDS.
    #pragma unroll
    for (int t = 0; t < 8; t++) {
      #pragma unroll
      for (int r = 0; r < 8; r++)
        Pl[wave][r + 8 * hi][t * 16 + l16] = (_Float16)s[t][r];
    }
    __syncthreads();

    // O += P * V. K-dim = 128 keys = 4 chunks; 4 n-subtiles of Dout.
    #pragma unroll
    for (int c = 0; c < 4; c++) {
      const _Float16* pp = &Pl[wave][l16][c * 32 + hi * 8];
      v16h ap = make_a16(*(const v8h*)pp, *(const v8h*)(pp + 16));
      #pragma unroll
      for (int n = 0; n < 4; n++) {
        const _Float16* bp = &VT[n * 16 + l16][c * 32 + hi * 8];
        v16h bv = make_a16(*(const v8h*)bp, *(const v8h*)(bp + 16));
        acc[n] = __builtin_amdgcn_wmma_f32_16x16x32_f16(
            false, ap, false, bv, (short)0, acc[n], false, false);
      }
    }
  }

  // Normalize and store f32 output.
  float* Orow = Out + (batch_base + qrow0) * DOUT;
  #pragma unroll
  for (int n = 0; n < 4; n++) {
    #pragma unroll
    for (int r = 0; r < 8; r++) {
      int row = r + 8 * hi;
      Orow[(size_t)row * DOUT + n * 16 + l16] = acc[n][r] / lrow[r];
    }
  }
}

extern "C" void kernel_launch(void* const* d_in, const int* in_sizes, int n_in,
                              void* d_out, int out_size, void* d_ws, size_t ws_size,
                              hipStream_t stream) {
  (void)in_sizes; (void)n_in; (void)out_size; (void)ws_size;
  const float* X = (const float*)d_in[0];   // [16,2048,512] f32
  const float* W = (const float*)d_in[1];   // [1000,512,64] f32

  // Workspace layout: [3][BS][64] f16 = 12.6 MB (Q, K, V).
  _Float16* ws = (_Float16*)d_ws;
  _Float16* Qh = ws;
  _Float16* Kh = ws + (size_t)1 * BS * DOUT;
  _Float16* Vh = ws + (size_t)2 * BS * DOUT;

  dim3 pg(BS / 128, 3);
  hipLaunchKernelGGL(proj_kernel, pg, dim3(256), 0, stream, X, W, ws);

  dim3 ag(SEQ / 128, BATCH);
  hipLaunchKernelGGL(attn_kernel, ag, dim3(256), 0, stream, Qh, Kh, Vh,
                     (float*)d_out);
}
